// GAT_36429912605263
// MI455X (gfx1250) — matrix operand order repaired and checked by reading, not compiled, not run on hardware
//
#include <hip/hip_runtime.h>

// ---------------- GAT on MI455X (gfx1250, wave32) ----------------
// Dense projections: fp32 -> (hi,lo) bf16 planes pre-split + pre-swizzled B,
// then a WMMA GEMM whose inner loop is pure b128 loads + v_wmma_f32_16x16x32_bf16
// (3 WMMAs per K-step per tile for ~fp32 accuracy; 4 col-tiles per wave).
// Graph softmax/aggregation via L2-resident float atomics.

typedef __attribute__((ext_vector_type(16))) __bf16 v16bf;
typedef __attribute__((ext_vector_type(8)))  __bf16 v8bf;
typedef __attribute__((ext_vector_type(8)))  float  v8f;

#define IN_F 256
#define HID  64
#define NH   4
#define CH   256  // NH * HID

__device__ __forceinline__ void splitbf(float f, __bf16 &hi, __bf16 &lo) {
  unsigned u = __builtin_bit_cast(unsigned, f);
  unsigned r = u + 0x7FFFu + ((u >> 16) & 1u);          // RNE to bf16
  unsigned short hs = (unsigned short)(r >> 16);
  hi = __builtin_bit_cast(__bf16, hs);
  float fh = __builtin_bit_cast(float, (unsigned)hs << 16);
  float rem = f - fh;                                   // residual
  unsigned u2 = __builtin_bit_cast(unsigned, rem);
  u2 += 0x7FFFu + ((u2 >> 16) & 1u);
  lo = __builtin_bit_cast(__bf16, (unsigned short)(u2 >> 16));
}

// Elementwise fp32 -> hi/lo bf16 planes (row-major preserved).
__global__ void split_hl(const float* __restrict__ X, __bf16* __restrict__ H,
                         __bf16* __restrict__ L, int count) {
  int i = blockIdx.x * blockDim.x + threadIdx.x;
  if (i >= count) return;
  __bf16 h, l;
  splitbf(X[i], h, l);
  H[i] = h; L[i] = l;
}

// Pack B[K,Nc] (row-major fp32) into WMMA B-fragment order, hi/lo planes:
// dest[((kt*2+half)*(Nc/16) + nTile)*256 + lane16*16 + e], k = kt*32+half*16+e.
__global__ void pack_B(const float* __restrict__ B, __bf16* __restrict__ PH,
                       __bf16* __restrict__ PL, int K, int Nc) {
  int i = blockIdx.x * blockDim.x + threadIdx.x;
  if (i >= K * Nc) return;
  int k = i / Nc, n = i % Nc;
  int kt = k >> 5, half = (k >> 4) & 1, e = k & 15;
  size_t d = ((size_t)((kt * 2 + half) * (Nc >> 4) + (n >> 4)) * 16 + (n & 15)) * 16 + e;
  __bf16 h, l;
  splitbf(B[i], h, l);
  PH[d] = h; PL[d] = l;
}

// C[M,Nc] = A @ B (+bias). One wave computes 16 rows x 64 cols (4 tiles).
// A planes row-major bf16; B planes pre-packed. M%16==0, K%32==0, Nc%64==0.
__global__ __launch_bounds__(128)
void wmma_gemm(const __bf16* __restrict__ AH, const __bf16* __restrict__ AL,
               const __bf16* __restrict__ BPH, const __bf16* __restrict__ BPL,
               const float* __restrict__ bias, float* __restrict__ C,
               int M, int K, int Nc) {
  const int tilesNG = Nc >> 6;                  // groups of 4 col-tiles
  const int tiles   = (M >> 4) * tilesNG;
  int w = (blockIdx.x << 2) + (threadIdx.x >> 5);
  if (w >= tiles) return;                       // wave-uniform; EXEC stays full
  const int tm = w / tilesNG, tng = w % tilesNG;
  const int lane = threadIdx.x & 31;
  const int l16 = lane & 15, half = lane >> 4;

  v8f acc0 = {}, acc1 = {}, acc2 = {}, acc3 = {};
  v8f* accs[4] = {&acc0, &acc1, &acc2, &acc3};

  // A frag: k=k0+half*16+e (e=0..15); both 16B chunks
  const __bf16* arow_h = AH + (size_t)(tm * 16 + l16) * K + half * 16;
  const __bf16* arow_l = AL + (size_t)(tm * 16 + l16) * K + half * 16;
  const int nt16 = Nc >> 4;

  for (int k0 = 0; k0 < K; k0 += 32) {
    v8bf ah0 = *(const v8bf*)(arow_h + k0);
    v8bf ah1 = *(const v8bf*)(arow_h + k0 + 8);
    v8bf al0 = *(const v8bf*)(arow_l + k0);
    v8bf al1 = *(const v8bf*)(arow_l + k0 + 8);
    v16bf ahf = __builtin_shufflevector(ah0, ah1, 0,1,2,3,4,5,6,7,8,9,10,11,12,13,14,15);
    v16bf alf = __builtin_shufflevector(al0, al1, 0,1,2,3,4,5,6,7,8,9,10,11,12,13,14,15);

    const size_t kbase = (size_t)(((k0 >> 5) * 2 + half) * nt16) * 256 + l16 * 16;
#pragma unroll
    for (int t = 0; t < 4; ++t) {
      const size_t boff = kbase + (size_t)(tng * 4 + t) * 256;
      v8bf bh0 = *(const v8bf*)(BPH + boff);
      v8bf bh1 = *(const v8bf*)(BPH + boff + 8);
      v8bf bl0 = *(const v8bf*)(BPL + boff);
      v8bf bl1 = *(const v8bf*)(BPL + boff + 8);
      v16bf bhf = __builtin_shufflevector(bh0, bh1, 0,1,2,3,4,5,6,7,8,9,10,11,12,13,14,15);
      v16bf blf = __builtin_shufflevector(bl0, bl1, 0,1,2,3,4,5,6,7,8,9,10,11,12,13,14,15);
      v8f a = *accs[t];
      a = __builtin_amdgcn_wmma_f32_16x16x32_bf16(false, alf, false, bhf, (short)0, a, false, false);
      a = __builtin_amdgcn_wmma_f32_16x16x32_bf16(false, ahf, false, blf, (short)0, a, false, false);
      a = __builtin_amdgcn_wmma_f32_16x16x32_bf16(false, ahf, false, bhf, (short)0, a, false, false);
      *accs[t] = a;
    }
  }

#pragma unroll
  for (int t = 0; t < 4; ++t) {
    const int col = (tng * 4 + t) * 16 + l16;
    const float bb = bias ? bias[col] : 0.f;
    float* crow = C + (size_t)(tm * 16 + half * 8) * Nc + col;   // C/D: M = v + half*8
    v8f a = *accs[t];
#pragma unroll
    for (int v = 0; v < 8; ++v) crow[(size_t)v * Nc] = a[v] + bb;
  }
}

// el/er: one wave per (node, head); 64-length dot via 2 loads/lane + shuffle tree.
__global__ void attn_elr(const float* __restrict__ feat, const float* __restrict__ al,
                         const float* __restrict__ ar, float* __restrict__ EL,
                         float* __restrict__ ER, int nh_tot) {
  int w = (blockIdx.x * blockDim.x + threadIdx.x) >> 5;
  if (w >= nh_tot) return;
  int lane = threadIdx.x & 31;
  int n = w >> 2, h = w & 3;
  const float* fr = feat + (size_t)n * CH + h * HID;
  const float* ap = al + h * HID;
  const float* rp = ar + h * HID;
  float f0 = fr[lane], f1 = fr[lane + 32];
  float sl = f0 * ap[lane] + f1 * ap[lane + 32];
  float sr = f0 * rp[lane] + f1 * rp[lane + 32];
#pragma unroll
  for (int off = 16; off > 0; off >>= 1) {
    sl += __shfl_xor(sl, off, 32);
    sr += __shfl_xor(sr, off, 32);
  }
  if (lane == 0) { EL[w] = sl; ER[w] = sr; }
}

__global__ void attn_init(float* __restrict__ out, const float* __restrict__ bias,
                          float* __restrict__ MX, float* __restrict__ SM, int Nn) {
  int idx = blockIdx.x * blockDim.x + threadIdx.x;
  if (idx < Nn * CH) out[idx] = bias[idx & (CH - 1)];
  if (idx < Nn * NH) { MX[idx] = __builtin_bit_cast(float, 0xFF800000u); SM[idx] = 0.f; }
}

__device__ __forceinline__ float lrelu(float v) { return v > 0.f ? v : 0.2f * v; }

// monotone int/uint bit trick for float atomic max (MX preset to -inf)
__device__ __forceinline__ void atomicMaxF(float* a, float v) {
  if (v >= 0.f) atomicMax((int*)a, __float_as_int(v));
  else          atomicMin((unsigned int*)a, __float_as_uint(v));
}

__global__ void edge_max(const int* __restrict__ src, const int* __restrict__ dst,
                         const float* __restrict__ EL, const float* __restrict__ ER,
                         float* __restrict__ MX, int Ecnt) {
  int e = blockIdx.x * blockDim.x + threadIdx.x;
  if (e >= Ecnt) return;
  int s = src[e], d = dst[e];
#pragma unroll
  for (int h = 0; h < NH; ++h)
    atomicMaxF(MX + d * NH + h, lrelu(EL[s * NH + h] + ER[d * NH + h]));
}

__global__ void edge_sum(const int* __restrict__ src, const int* __restrict__ dst,
                         const float* __restrict__ EL, const float* __restrict__ ER,
                         const float* __restrict__ MX, float* __restrict__ SM, int Ecnt) {
  int e = blockIdx.x * blockDim.x + threadIdx.x;
  if (e >= Ecnt) return;
  int s = src[e], d = dst[e];
#pragma unroll
  for (int h = 0; h < NH; ++h) {
    float v = lrelu(EL[s * NH + h] + ER[d * NH + h]);
    atomicAdd(SM + d * NH + h, expf(v - MX[d * NH + h]));
  }
}

// one 256-thread block per edge: tid t -> head t>>6, feature t&63 (h*64+f == t)
__global__ void edge_agg(const int* __restrict__ src, const int* __restrict__ dst,
                         const float* __restrict__ EL, const float* __restrict__ ER,
                         const float* __restrict__ MX, const float* __restrict__ SM,
                         const float* __restrict__ feat, float* __restrict__ out) {
  int e = blockIdx.x, t = threadIdx.x;
  int s = src[e], d = dst[e];
  int h = t >> 6;
  float v = lrelu(EL[s * NH + h] + ER[d * NH + h]);
  float alpha = expf(v - MX[d * NH + h]) / SM[d * NH + h];
  atomicAdd(out + (size_t)d * CH + t, alpha * feat[(size_t)s * CH + t]);
}

__global__ void relu_ip(float* __restrict__ x, int n) {
  int i = blockIdx.x * blockDim.x + threadIdx.x;
  if (i < n) x[i] = fmaxf(x[i], 0.f);
}

// out[n] = sigmoid(dot(P[n,0:64], w) + b): one wave per node.
__global__ void final_mlp(const float* __restrict__ P, const float* __restrict__ w,
                          const float* __restrict__ b, float* __restrict__ out, int Nn) {
  int n = (blockIdx.x * blockDim.x + threadIdx.x) >> 5;
  if (n >= Nn) return;
  int lane = threadIdx.x & 31;
  const float* pr = P + (size_t)n * HID;
  float s = pr[lane] * w[lane] + pr[lane + 32] * w[lane + 32];
#pragma unroll
  for (int off = 16; off > 0; off >>= 1) s += __shfl_xor(s, off, 32);
  if (lane == 0) out[n] = 1.f / (1.f + expf(-(s + b[0])));
}

extern "C" void kernel_launch(void* const* d_in, const int* in_sizes, int n_in,
                              void* d_out, int out_size, void* d_ws, size_t ws_size,
                              hipStream_t stream) {
  const float* x   = (const float*)d_in[0];
  const int*   src = (const int*)d_in[1];
  const int*   dst = (const int*)d_in[2];
  const float* eW  = (const float*)d_in[3];
  const float* eb  = (const float*)d_in[4];
  const float* W1  = (const float*)d_in[5];
  const float* al1 = (const float*)d_in[6];
  const float* ar1 = (const float*)d_in[7];
  const float* b1  = (const float*)d_in[8];
  const float* W2  = (const float*)d_in[9];
  const float* al2 = (const float*)d_in[10];
  const float* ar2 = (const float*)d_in[11];
  const float* b2  = (const float*)d_in[12];
  const float* p1W = (const float*)d_in[13];
  const float* p1b = (const float*)d_in[14];
  const float* p2W = (const float*)d_in[15];
  const float* p2b = (const float*)d_in[16];
  float* out = (float*)d_out;

  const int Nn   = in_sizes[0] / IN_F;   // 50000 (divisible by 16)
  const int Ecnt = in_sizes[1];          // 500000

  // bf16 planes first (16B-aligned for any even Nn), then fp32 buffers.
  const size_t aElems = (size_t)Nn * CH;          // max A elements per GEMM
  const size_t bElems = (size_t)CH * CH;          // max B elements
  size_t need = aElems * 2 * 2 + bElems * 2 * 2
              + (size_t)Nn * (HID + CH + CH + 4 * NH) * sizeof(float);
  if (ws_size < need) return;

  __bf16* AH  = (__bf16*)d_ws;           // [Nn*256] hi plane of current A
  __bf16* ALo = AH + aElems;             // [Nn*256] lo plane
  __bf16* BPH = ALo + aElems;            // packed B hi (<= 256*256)
  __bf16* BPL = BPH + bElems;            // packed B lo
  float*  B0  = (float*)(BPL + bElems);  // [Nn,64]  embed out, later p1 out
  float*  BF  = B0 + (size_t)Nn * HID;   // [Nn,256] feat1 / feat2
  float*  BO  = BF + (size_t)Nn * CH;    // [Nn,256] out1 / out2
  float*  EL  = BO + (size_t)Nn * CH;    // [Nn,4]
  float*  ER  = EL + (size_t)Nn * NH;
  float*  MX  = ER + (size_t)Nn * NH;
  float*  SM  = MX + (size_t)Nn * NH;

  auto gemm = [&](const float* A, const float* Bw, const float* bias, float* C,
                  int M, int K, int Nc) {
    int acnt = M * K, bcnt = K * Nc;
    split_hl<<<dim3((acnt + 255) / 256), dim3(256), 0, stream>>>(A, AH, ALo, acnt);
    pack_B<<<dim3((bcnt + 255) / 256), dim3(256), 0, stream>>>(Bw, BPH, BPL, K, Nc);
    int tiles = (M / 16) * (Nc / 64);
    wmma_gemm<<<dim3((tiles + 3) / 4), dim3(128), 0, stream>>>(AH, ALo, BPH, BPL, bias, C, M, K, Nc);
  };
  auto attn = [&](const float* feat, const float* al, const float* ar,
                  const float* bias, float* o) {
    int nh = Nn * NH;
    attn_elr<<<dim3((nh + 7) / 8), dim3(256), 0, stream>>>(feat, al, ar, EL, ER, nh);
    attn_init<<<dim3((Nn * CH + 255) / 256), dim3(256), 0, stream>>>(o, bias, MX, SM, Nn);
    edge_max<<<dim3((Ecnt + 255) / 256), dim3(256), 0, stream>>>(src, dst, EL, ER, MX, Ecnt);
    edge_sum<<<dim3((Ecnt + 255) / 256), dim3(256), 0, stream>>>(src, dst, EL, ER, MX, SM, Ecnt);
    edge_agg<<<dim3(Ecnt), dim3(256), 0, stream>>>(src, dst, EL, ER, MX, SM, feat, o);
    relu_ip<<<dim3((Nn * CH + 255) / 256), dim3(256), 0, stream>>>(o, Nn * CH);
  };

  gemm(x, eW, eb, B0, Nn, IN_F, HID);        // embed: h0 = x@embed_W + b
  gemm(B0, W1, nullptr, BF, Nn, HID, CH);    // feat1 = h0@W1
  attn(BF, al1, ar1, b1, BO);                // out1  = relu(GAT1)
  gemm(BO, W2, nullptr, BF, Nn, CH, CH);     // feat2 = out1@W2
  attn(BF, al2, ar2, b2, BO);                // out2  = relu(GAT2)
  gemm(BO, p1W, p1b, B0, Nn, CH, HID);       // p1    = out2@p1_W + p1_b
  final_mlp<<<dim3((Nn + 7) / 8), dim3(256), 0, stream>>>(B0, p2W, p2b, out, Nn);
}
